// LoRALinear_14499809591526
// MI455X (gfx1250) — matrix operand order repaired and checked by reading, not compiled, not run on hardware
//
#include <hip/hip_runtime.h>
#include <stdint.h>

#define K_DIM 4096
#define N_DIM 4096
#define R_DIM 16
#define BM 128
#define BN 128
#define BK 32
#define SCALING 2.0f

typedef __attribute__((ext_vector_type(16))) __bf16 v16bf;
typedef __attribute__((ext_vector_type(8)))  __bf16 v8bf;
typedef __attribute__((ext_vector_type(8)))  float  v8f;

union V16 { v16bf v; v8bf h[2]; };

// Split an f32 into bf16 hi (truncated top 16 bits) and bf16 lo (residual).
// hi + lo carries ~16 mantissa bits -> near-f32 accuracy from bf16 WMMAs.
__device__ __forceinline__ void split2(float f, unsigned short& hi, unsigned short& lo) {
    unsigned int u = __float_as_uint(f);
    unsigned short h = (unsigned short)(u >> 16);
    float fh = __uint_as_float(((unsigned int)h) << 16);
    float r  = f - fh;
    hi = h;
    lo = (unsigned short)(__float_as_uint(r) >> 16);
}

__device__ __forceinline__ void split4(const float4 f, ushort4& hi, ushort4& lo) {
    split2(f.x, hi.x, lo.x);
    split2(f.y, hi.y, lo.y);
    split2(f.z, hi.z, lo.z);
    split2(f.w, hi.w, lo.w);
}

// Async DMA of 16 bytes global -> LDS (GV mode, per-lane), tracked by ASYNCcnt.
// LDS destination offset comes from the low 32 bits of the generic pointer
// (flat->LDS mapping truncates addr[31:0] = wave-relative LDS byte offset).
__device__ __forceinline__ void async_copy16(const void* gptr, unsigned lds_off) {
    asm volatile("global_load_async_to_lds_b128 %0, %1, off"
                 :: "v"(lds_off), "v"(gptr) : "memory");
}

__device__ __forceinline__ void wait_async0() {
#if __has_builtin(__builtin_amdgcn_s_wait_asynccnt)
    __builtin_amdgcn_s_wait_asynccnt(0);
#else
    asm volatile("s_wait_asynccnt 0" ::: "memory");
#endif
}

// ---------------------------------------------------------------------------
// Kernel 1: t[m, r] = sum_k x[m,k] * A[r,k]   (t: M x 16)
// One block (256 threads = 16 r-lanes x 16 k-groups) per row m.
// ---------------------------------------------------------------------------
__global__ __launch_bounds__(256)
void lora_xa_kernel(const float* __restrict__ X, const float* __restrict__ A,
                    float* __restrict__ T) {
    __shared__ float red[16][17];
    const int m = blockIdx.x;
    const int r = threadIdx.x & 15;
    const int g = threadIdx.x >> 4;

    const float* xrow = X + (size_t)m * K_DIM;
    const float* arow = A + (size_t)r * K_DIM;

    float acc = 0.f;
    for (int k = g * 4; k < K_DIM; k += 64) {
        float4 xv = *(const float4*)(xrow + k);
        float4 av = *(const float4*)(arow + k);
        acc += xv.x * av.x + xv.y * av.y + xv.z * av.z + xv.w * av.w;
    }
    red[g][r] = acc;
    __syncthreads();
    if (threadIdx.x < 16) {
        float s = 0.f;
        #pragma unroll
        for (int i = 0; i < 16; ++i) s += red[i][threadIdx.x];
        T[(size_t)m * R_DIM + threadIdx.x] = s;
    }
}

// ---------------------------------------------------------------------------
// Kernel 2: out[m,n] = sum_k x[m,k]*W[n,k] + b[n] + 2 * sum_r t[m,r]*B[n,r]
// Block tile 128x128, 8 wave32s (2 in M x 4 in N), wave tile 64x32.
// Double-buffered async global->LDS staging (ASYNCcnt), then split-bf16
// (hi/lo) GEMM: 3x v_wmma_f32_16x16x32_bf16 per 16x16x32 step.
// ---------------------------------------------------------------------------
__global__ __launch_bounds__(256)
void lora_gemm_kernel(const float* __restrict__ X, const float* __restrict__ Wm,
                      const float* __restrict__ bias, const float* __restrict__ Bm,
                      const float* __restrict__ T, float* __restrict__ Out) {
    __shared__ __align__(16) float          xf[2][BM][BK];     // raw f32 X tiles
    __shared__ __align__(16) float          wf[2][BN][BK];     // raw f32 W tiles
    __shared__ __align__(16) unsigned short xs_hi[BM][BK];
    __shared__ __align__(16) unsigned short xs_lo[BM][BK];
    __shared__ __align__(16) unsigned short ws_hi[BN][BK];
    __shared__ __align__(16) unsigned short ws_lo[BN][BK];
    __shared__ __align__(16) float ts[BM][R_DIM];   // t rows for this M-tile
    __shared__ __align__(16) float bls[BN][R_DIM];  // B rows for this N-tile

    const int tid     = threadIdx.x;
    const int lane    = tid & 31;
    const int wave    = tid >> 5;
    const int waveM   = wave >> 2;  // 0..1  -> 64-row slab
    const int waveN   = wave & 3;   // 0..3  -> 32-col slab
    const int lrow    = lane & 15;
    const int halfsel = lane >> 4;  // 0 or 1 (ISA lane-half K selector)

    const int Mbase = blockIdx.y * BM;
    const int Nbase = blockIdx.x * BN;

    // This thread's 4 staging chunks (16B each): row/col fixed for all tiles.
    int srow[4], skc[4];
    #pragma unroll
    for (int i = 0; i < 4; ++i) {
        const int c = tid + 256 * i;   // float4 chunk id, 0..1023
        srow[i] = c >> 3;              // 0..127
        skc[i]  = (c & 7) * 4;         // 0,4,...,28
    }

    v8f acc[4][2];
    #pragma unroll
    for (int mi = 0; mi < 4; ++mi)
        #pragma unroll
        for (int nj = 0; nj < 2; ++nj)
            acc[mi][nj] = {};

    const int NT = K_DIM / BK;

    // Prologue: kick off async DMA of tile 0 into buffer 0.
    #pragma unroll
    for (int i = 0; i < 4; ++i) {
        async_copy16(X  + (size_t)(Mbase + srow[i]) * K_DIM + skc[i],
                     (unsigned)(uintptr_t)&xf[0][srow[i]][skc[i]]);
        async_copy16(Wm + (size_t)(Nbase + srow[i]) * K_DIM + skc[i],
                     (unsigned)(uintptr_t)&wf[0][srow[i]][skc[i]]);
    }

    for (int kt = 0; kt < NT; ++kt) {
        const int cur = kt & 1;
        const int nxt = cur ^ 1;

        // Current f32 tile landed; previous iteration's fragment reads done.
        wait_async0();
        __syncthreads();

        // Stream next tile into the other buffer while we convert + compute.
        if (kt + 1 < NT) {
            const int k0n = (kt + 1) * BK;
            #pragma unroll
            for (int i = 0; i < 4; ++i) {
                async_copy16(X  + (size_t)(Mbase + srow[i]) * K_DIM + k0n + skc[i],
                             (unsigned)(uintptr_t)&xf[nxt][srow[i]][skc[i]]);
                async_copy16(Wm + (size_t)(Nbase + srow[i]) * K_DIM + k0n + skc[i],
                             (unsigned)(uintptr_t)&wf[nxt][srow[i]][skc[i]]);
            }
        }

        // Convert f32 LDS tile -> split bf16 hi/lo planes.
        #pragma unroll
        for (int i = 0; i < 4; ++i) {
            const int row = srow[i], kc = skc[i];
            float4 xv = *(const float4*)&xf[cur][row][kc];
            float4 wv = *(const float4*)&wf[cur][row][kc];
            ushort4 xh, xl, wh, wl;
            split4(xv, xh, xl);
            split4(wv, wh, wl);
            *(ushort4*)&xs_hi[row][kc] = xh;
            *(ushort4*)&xs_lo[row][kc] = xl;
            *(ushort4*)&ws_hi[row][kc] = wh;
            *(ushort4*)&ws_lo[row][kc] = wl;
        }
        __syncthreads();

        // B fragments (W tile): lane-half selects K 0..15 / 16..31, contiguous.
        V16 bh[2], bl[2];
        #pragma unroll
        for (int nj = 0; nj < 2; ++nj) {
            const int nrow = waveN * 32 + nj * 16 + lrow;
            const int kc   = halfsel * 16;
            bh[nj].h[0] = *(const v8bf*)&ws_hi[nrow][kc];
            bh[nj].h[1] = *(const v8bf*)&ws_hi[nrow][kc + 8];
            bl[nj].h[0] = *(const v8bf*)&ws_lo[nrow][kc];
            bl[nj].h[1] = *(const v8bf*)&ws_lo[nrow][kc + 8];
        }

        #pragma unroll
        for (int mi = 0; mi < 4; ++mi) {
            // A fragments: lane-half selects K 0..15 / 16..31, contiguous.
            const int mrow = waveM * 64 + mi * 16 + lrow;
            const int kc   = halfsel * 16;
            V16 ah, al;
            ah.h[0] = *(const v8bf*)&xs_hi[mrow][kc];
            ah.h[1] = *(const v8bf*)&xs_hi[mrow][kc + 8];
            al.h[0] = *(const v8bf*)&xs_lo[mrow][kc];
            al.h[1] = *(const v8bf*)&xs_lo[mrow][kc + 8];
            #pragma unroll
            for (int nj = 0; nj < 2; ++nj) {
                acc[mi][nj] = __builtin_amdgcn_wmma_f32_16x16x32_bf16(
                    false, ah.v, false, bh[nj].v, (short)0, acc[mi][nj], false, false);
                acc[mi][nj] = __builtin_amdgcn_wmma_f32_16x16x32_bf16(
                    false, ah.v, false, bl[nj].v, (short)0, acc[mi][nj], false, false);
                acc[mi][nj] = __builtin_amdgcn_wmma_f32_16x16x32_bf16(
                    false, al.v, false, bh[nj].v, (short)0, acc[mi][nj], false, false);
            }
        }
    }

    // Epilogue: stage t rows (M-tile) and B rows (N-tile) into LDS.
    __syncthreads();
    #pragma unroll
    for (int i = 0; i < 2; ++i) {
        const int c   = tid + 256 * i;  // 0..511 float4 chunks
        const int row = c >> 2;         // 0..127
        const int rc  = (c & 3) * 4;    // 0,4,8,12
        *(float4*)&ts[row][rc]  = *(const float4*)(T  + (size_t)(Mbase + row) * R_DIM + rc);
        *(float4*)&bls[row][rc] = *(const float4*)(Bm + (size_t)(Nbase + row) * R_DIM + rc);
    }
    __syncthreads();

    // C/D layout (ISA 7.12.2): VGPR r -> M = r (lanes 0-15) / r+8 (lanes 16-31),
    // N = lane%16.
    #pragma unroll
    for (int nj = 0; nj < 2; ++nj) {
        const int nloc = waveN * 32 + nj * 16 + lrow;
        const int n    = Nbase + nloc;
        const float bv = bias[n];
        #pragma unroll
        for (int mi = 0; mi < 4; ++mi) {
            #pragma unroll
            for (int r = 0; r < 8; ++r) {
                const int mloc = waveM * 64 + mi * 16 + halfsel * 8 + r;
                float dot = 0.f;
                #pragma unroll
                for (int rr = 0; rr < 16; ++rr)
                    dot += ts[mloc][rr] * bls[nloc][rr];
                Out[(size_t)(Mbase + mloc) * N_DIM + n] =
                    acc[mi][nj][r] + bv + SCALING * dot;
            }
        }
    }
}

extern "C" void kernel_launch(void* const* d_in, const int* in_sizes, int n_in,
                              void* d_out, int out_size, void* d_ws, size_t ws_size,
                              hipStream_t stream) {
    const float* X    = (const float*)d_in[0];  // (M, 4096)
    const float* Wm   = (const float*)d_in[1];  // (4096, 4096)
    const float* bias = (const float*)d_in[2];  // (4096,)
    const float* A    = (const float*)d_in[3];  // (16, 4096)
    const float* Bm   = (const float*)d_in[4];  // (4096, 16)
    float* Out = (float*)d_out;
    float* T   = (float*)d_ws;                  // (M, 16) scratch

    const int M = in_sizes[0] / K_DIM;

    lora_xa_kernel<<<M, 256, 0, stream>>>(X, A, T);

    dim3 grid(N_DIM / BN, M / BM);
    lora_gemm_kernel<<<grid, 256, 0, stream>>>(X, Wm, bias, Bm, T, Out);
}
